// GraphAttentionLayer_54185307406961
// MI455X (gfx1250) — matrix operand order repaired and checked by reading, not compiled, not run on hardware
//
#include <hip/hip_runtime.h>

typedef _Float16 half8 __attribute__((ext_vector_type(8)));
typedef _Float16 h4    __attribute__((ext_vector_type(4)));
typedef _Float16 v16h  __attribute__((ext_vector_type(16)));
typedef float    v8f   __attribute__((ext_vector_type(8)));
typedef float    f4    __attribute__((ext_vector_type(4)));

#define NN     6144
#define FI     256
#define FO     64
#define LALPHA 0.2f
#define NEGC   -9e15f
#define ROWS   16
#define JTILE  256
#define AT_STRIDE 264      // 256 + 8 halfs pad -> conflict-free b128 LDS reads
#define MWORDS (NN / 16)   // 384 x uint16 mask words per row

// cheap online-softmax push (1 exp in the common path); matches reference
// semantics for masked(-9e15) and all-masked rows exactly
__device__ __forceinline__ void sm_push(float& m, float& l, float e) {
  if (e <= m) {
    l += __expf(e - m);
  } else {
    l = l * __expf(m - e) + 1.f;
    m = e;
  }
}
__device__ __forceinline__ void sm_combine(float& m, float& l, float mo, float lo) {
  float mn = fmaxf(m, mo);
  l = l * __expf(m - mn) + lo * __expf(mo - mn);
  m = mn;
}

// Wh = h @ W  (f32), plus transposed f16 copy WhT[n][i] for WMMA B-fragments
__global__ void k_gemm_wh(const float* __restrict__ h, const float* __restrict__ W,
                          float* __restrict__ Wh, _Float16* __restrict__ WhT) {
  int idx = blockIdx.x * blockDim.x + threadIdx.x;   // NN*FO threads
  int i = idx >> 6, n = idx & 63;
  float acc = 0.f;
  const float* hr = h + (long)i * FI;
  #pragma unroll 4
  for (int k = 0; k < FI; ++k) acc = fmaf(hr[k], W[k * FO + n], acc);
  Wh[idx] = acc;
  WhT[(long)n * NN + i] = (_Float16)acc;
}

// s1 = Wh @ a[:FO], s2 = Wh @ a[FO:]
__global__ void k_scores(const float* __restrict__ Wh, const float* __restrict__ a,
                         float* __restrict__ s1, float* __restrict__ s2) {
  int i = blockIdx.x * blockDim.x + threadIdx.x;
  const float* r = Wh + (long)i * FO;
  float x = 0.f, y = 0.f;
  #pragma unroll
  for (int f = 0; f < FO; ++f) { x = fmaf(r[f], a[f], x); y = fmaf(r[f], a[FO + f], y); }
  s1[i] = x; s2[i] = y;
}

// Fused: masked softmax -> mix with M -> masked softmax -> (attention @ Wh) via WMMA -> elu
__global__ void __launch_bounds__(256) k_attn(
    const int* __restrict__ adj, const float* __restrict__ M,
    const float* __restrict__ aw, const float* __restrict__ s1,
    const float* __restrict__ s2, const _Float16* __restrict__ WhT,
    float* __restrict__ out) {
  __shared__ _Float16 att[ROWS * AT_STRIDE];
  __shared__ float red[8][ROWS][FO];
  __shared__ unsigned short mask16[ROWS * MWORDS];   // 12 KB adjacency bitmask
  __shared__ float rs_s1[ROWS], rs_m1[ROWS], rs_i1[ROWS], rs_m2[ROWS], rs_i2[ROWS];

  const int t  = threadIdx.x;
  const int i0 = blockIdx.x * ROWS;

  // softmax(aw) -> mixing weights
  float awm = fmaxf(aw[0], aw[1]);
  float e0 = __expf(aw[0] - awm), e1 = __expf(aw[1] - awm);
  float w0 = e0 / (e0 + e1), w1 = e1 / (e0 + e1);

  if (t < ROWS) rs_s1[t] = s1[i0 + t];
  __syncthreads();

  const int r = t >> 4, c = t & 15;            // 16 threads per row (wave32 halves)
  const long rowbase = (long)(i0 + r) * NN;
  const float s1r = rs_s1[r];

  // ---- pass 1: softmax-1 stats; capture adjacency bitmask via wave32 ballot ----
  float m = NEGC, l = 0.f;
  for (int jb = 0; jb < NN; jb += 16) {
    int j = jb + c;
    bool mk = adj[rowbase + j] > 0;
    unsigned bal = __builtin_amdgcn_ballot_w32(mk);  // lanes 0-15: row r(even), 16-31: row r(odd)
    if (c == 0)
      mask16[r * MWORDS + (jb >> 4)] =
          (unsigned short)(((t & 31) < 16) ? (bal & 0xFFFFu) : (bal >> 16));
    float x = s1r + s2[j];
    float e = (x > 0.f) ? x : LALPHA * x;
    e = mk ? e : NEGC;
    sm_push(m, l, e);
  }
  for (int off = 1; off < 16; off <<= 1) {
    float mo = __shfl_xor(m, off, 16);
    float lo = __shfl_xor(l, off, 16);
    sm_combine(m, l, mo, lo);
  }
  if (c == 0) { rs_m1[r] = m; rs_i1[r] = 1.f / l; }
  __syncthreads();

  // ---- pass 2: softmax-2 stats over w0*p + w1*M (mask from LDS, M via b128) ----
  const float m1 = rs_m1[r], i1 = rs_i1[r];
  const int c4 = c * 4;
  m = NEGC; l = 0.f;
  for (int jb = 0; jb < NN; jb += 64) {
    int j = jb + c4;
    f4 Mv  = *(const f4*)(M + rowbase + j);
    f4 s2v = *(const f4*)(s2 + j);
    unsigned bits = mask16[r * MWORDS + (j >> 4)] >> (c4 & 15);
    #pragma unroll
    for (int q = 0; q < 4; ++q) {
      float x = s1r + s2v[q];
      float e = (x > 0.f) ? x : LALPHA * x;
      float p = __expf(e - m1) * i1;
      float v = ((bits >> q) & 1u) ? fmaf(w0, p, w1 * Mv[q]) : NEGC;
      sm_push(m, l, v);
    }
  }
  for (int off = 1; off < 16; off <<= 1) {
    float mo = __shfl_xor(m, off, 16);
    float lo = __shfl_xor(l, off, 16);
    sm_combine(m, l, mo, lo);
  }
  if (c == 0) { rs_m2[r] = m; rs_i2[r] = 1.f / l; }
  __syncthreads();

  // ---- pass 3: stage attention tiles in LDS, WMMA-accumulate att @ Wh ----
  const int lane = t & 31, wave = t >> 5;
  const int kbase = wave * 32;                 // each wave owns a 32-wide K chunk
  const int acol  = lane & 15;
  const int ccol  = (t & 63) * 4;              // staging: 4 cols per thread
  const int rgrp  = t >> 6;                    // staging: row group 0..3
  v8f cacc[4] = {};

  for (int j0 = 0; j0 < NN; j0 += JTILE) {
    __syncthreads();                           // prior tile fully consumed
    f4 s2v = *(const f4*)(s2 + j0 + ccol);     // invariant across sub-rows
    #pragma unroll
    for (int sub = 0; sub < 4; ++sub) {
      int row = sub * 4 + rgrp;
      long rb = (long)(i0 + row) * NN;
      f4 Mv = *(const f4*)(M + rb + j0 + ccol);
      unsigned bits = mask16[row * MWORDS + ((j0 + ccol) >> 4)] >> (ccol & 15);
      float sA = rs_s1[row], mA = rs_m1[row], iA = rs_i1[row];
      float mB = rs_m2[row], iB = rs_i2[row];
      h4 av;
      #pragma unroll
      for (int q = 0; q < 4; ++q) {
        float x = sA + s2v[q];
        float e = (x > 0.f) ? x : LALPHA * x;
        float p = __expf(e - mA) * iA;
        float v = ((bits >> q) & 1u) ? fmaf(w0, p, w1 * Mv[q]) : NEGC;
        av[q] = (_Float16)(__expf(v - mB) * iB);   // masked -> exactly 0
      }
      *(h4*)(att + row * AT_STRIDE + ccol) = av;
    }
    __syncthreads();

    // A fragment: 16x32 f16 layout (ISA 7.12.2): lanes<16 K=0..7/16..23, lanes>=16 K=8..15/24..31
    const _Float16* ar = att + acol * AT_STRIDE + kbase;
    half8 alo = *(const half8*)(ar + ((lane < 16) ? 0 : 8));
    half8 ahi = *(const half8*)(ar + ((lane < 16) ? 16 : 24));
    v16h afrag;
    #pragma unroll
    for (int q = 0; q < 8; ++q) { afrag[q] = alo[q]; afrag[q + 8] = ahi[q]; }

    // B fragment: 32x16, lane n=acol, K=(lane<16?0:8)+e / +16 ; WhT is [FO][NN] so K contiguous
    const int jrow = j0 + kbase + ((lane < 16) ? 0 : 8);
    #pragma unroll
    for (int nt = 0; nt < 4; ++nt) {
      const _Float16* bp = WhT + (long)(nt * 16 + acol) * NN + jrow;
      half8 blo = *(const half8*)(bp);
      half8 bhi = *(const half8*)(bp + 16);
      v16h bfrag;
      #pragma unroll
      for (int q = 0; q < 8; ++q) { bfrag[q] = blo[q]; bfrag[q + 8] = bhi[q]; }
      cacc[nt] = __builtin_amdgcn_wmma_f32_16x16x32_f16(
          false, afrag, false, bfrag, (short)0, cacc[nt], false, false);
    }
  }

  // ---- reduce wave partials, elu, store ----
  #pragma unroll
  for (int nt = 0; nt < 4; ++nt)
    #pragma unroll
    for (int v = 0; v < 8; ++v) {
      int mrow = v + ((lane < 16) ? 0 : 8);     // C layout: VGPR v, lane halves
      red[wave][mrow][nt * 16 + acol] = cacc[nt][v];
    }
  __syncthreads();
  #pragma unroll
  for (int q = 0; q < 4; ++q) {
    int idx = q * 256 + t;
    int mrow = idx >> 6, nn = idx & 63;
    float s = 0.f;
    #pragma unroll
    for (int wv = 0; wv < 8; ++wv) s += red[wv][mrow][nn];
    out[(long)(i0 + mrow) * FO + nn] = (s > 0.f) ? s : expm1f(s);
  }
}

extern "C" void kernel_launch(void* const* d_in, const int* in_sizes, int n_in,
                              void* d_out, int out_size, void* d_ws, size_t ws_size,
                              hipStream_t stream) {
  (void)in_sizes; (void)n_in; (void)out_size; (void)ws_size;
  const float* h   = (const float*)d_in[0];
  const int*   adj = (const int*)d_in[1];
  const float* M   = (const float*)d_in[2];
  const float* W   = (const float*)d_in[3];
  const float* a   = (const float*)d_in[4];
  const float* aw  = (const float*)d_in[5];
  float* out = (float*)d_out;

  // workspace layout (16B-aligned): Wh f32 | WhT f16 | s1 | s2  (~2.4 MB)
  float*    Wh  = (float*)d_ws;
  _Float16* WhT = (_Float16*)(Wh + (long)NN * FO);
  float*    s1  = (float*)(WhT + (long)NN * FO);
  float*    s2  = s1 + NN;

  k_gemm_wh<<<(NN * FO) / 256, 256, 0, stream>>>(h, W, Wh, WhT);
  k_scores<<<NN / 256, 256, 0, stream>>>(Wh, a, s1, s2);
  k_attn<<<NN / ROWS, 256, 0, stream>>>(adj, M, aw, s1, s2, WhT, out);
}
